// SingleHeadDilatedSelfAttention_43585328120520
// MI455X (gfx1250) — matrix operand order repaired and checked by reading, not compiled, not run on hardware
//
#include <hip/hip_runtime.h>
#include <hip/hip_bf16.h>

// ---------------- types ----------------
typedef _Float16 half4  __attribute__((ext_vector_type(4)));
typedef _Float16 half8  __attribute__((ext_vector_type(8)));
typedef _Float16 v16h   __attribute__((ext_vector_type(16)));
typedef float    v8f    __attribute__((ext_vector_type(8)));
typedef float    f4     __attribute__((ext_vector_type(4)));

// ---------------- problem constants ----------------
#define BATCH 4
#define TLEN  8192
#define EDIM  1024
#define DK    128
#define SEG   64        // positions per segment (j)
#define NSD   64        // dilated segment count (l / o / d indices)
#define NORMF 0.08838834764831845f   // 1/sqrt(128)
#define BIGNEG (-10000.0f)

// ---------------- LDS layout (halves / floats), padded pitches ----------------
#define XP 1032          // X pitch in halves  (2064B/row -> 516 dw, %64=4: conflict-free frag loads)
#define QP 136           // Q/K pitch in halves (272B -> 68 dw, %64=4)
#define VP 72            // Vt pitch in halves  (144B -> 36 dw)
#define SP 68            // S  pitch in floats  (272B -> 68 dw, %64=4)
#define HP 72            // softmaxed-S pitch in halves (144B -> 36 dw)

#define OFF_X  0
#define OFF_Q  132096                    // 64*1032*2
#define OFF_K  (OFF_Q + 17408)           // 64*136*2
#define OFF_VT (OFF_K + 17408)
#define OFF_SF (OFF_VT + 18432)          // 128*72*2
#define OFF_SH (OFF_SF + 17408)          // 64*68*4
#define LDS_BYTES (OFF_SH + 9216)        // 64*72*2  -> 211968 B total

#define WMMA_F16(a, b, c) \
    __builtin_amdgcn_wmma_f32_16x16x32_f16(false, (a), false, (b), (short)0, (c), false, false)

// ---------------- fragment loaders (per CDNA5 ISA layouts) ----------------
// A-matrix 16x32 f16: lane<16 -> M=lane, halves = K[0..7],K[16..23]; lane>=16 -> K[8..15],K[24..31]
__device__ __forceinline__ v16h ldsA(const _Float16* base, int pitch, int row0, int k0) {
    int lane = threadIdx.x & 31;
    const _Float16* p = base + (row0 + (lane & 15)) * pitch + k0 + ((lane >> 4) << 3);
    half8 lo = *(const half8*)(p);
    half8 hi = *(const half8*)(p + 16);
    return __builtin_shufflevector(lo, hi, 0,1,2,3,4,5,6,7,8,9,10,11,12,13,14,15);
}
// B-matrix 32x16 f16 from row-major storage M[n][k] (so B[k][n] = M[n][k]):
// lane<16 -> N=lane, halves=K[0..7],K[16..23]; lane>=16 -> N=lane-16, halves=K[8..15],K[24..31]
__device__ __forceinline__ v16h ldsB(const _Float16* base, int pitch, int n0, int k0) {
    int lane = threadIdx.x & 31;
    const _Float16* p = base + (n0 + (lane & 15)) * pitch + k0 + ((lane >> 4) << 3);
    half8 lo = *(const half8*)(p);
    half8 hi = *(const half8*)(p + 16);
    return __builtin_shufflevector(lo, hi, 0,1,2,3,4,5,6,7,8,9,10,11,12,13,14,15);
}
// A-fragment from global f32 row-major W[128][1024]: A[m][k] = W[m][k]
__device__ __forceinline__ v16h gblAW(const float* __restrict__ W, int m0, int k0) {
    int lane = threadIdx.x & 31;
    const float* p = W + (size_t)(m0 + (lane & 15)) * EDIM + k0 + ((lane >> 4) << 3);
    f4 a0 = *(const f4*)(p);
    f4 a1 = *(const f4*)(p + 4);
    f4 a2 = *(const f4*)(p + 16);
    f4 a3 = *(const f4*)(p + 20);
    v16h a;
#pragma unroll
    for (int i = 0; i < 4; ++i) {
        a[i]      = (_Float16)a0[i];
        a[4 + i]  = (_Float16)a1[i];
        a[8 + i]  = (_Float16)a2[i];
        a[12 + i] = (_Float16)a3[i];
    }
    return a;
}
// B-fragment from global f32 row-major W[128][1024]: B[k][m] = W[m][k]
__device__ __forceinline__ v16h gblBW(const float* __restrict__ W, int n0, int k0) {
    int lane = threadIdx.x & 31;
    const float* p = W + (size_t)(n0 + (lane & 15)) * EDIM + k0 + ((lane >> 4) << 3);
    f4 a0 = *(const f4*)(p);
    f4 a1 = *(const f4*)(p + 4);
    f4 a2 = *(const f4*)(p + 16);
    f4 a3 = *(const f4*)(p + 20);
    v16h b;
#pragma unroll
    for (int i = 0; i < 4; ++i) {
        b[i]      = (_Float16)a0[i];
        b[4 + i]  = (_Float16)a1[i];
        b[8 + i]  = (_Float16)a2[i];
        b[12 + i] = (_Float16)a3[i];
    }
    return b;
}

__global__ __launch_bounds__(256)
void dilated_attn_kernel(const float* __restrict__ x,
                         const float* __restrict__ P,
                         const float* __restrict__ Wk,
                         const float* __restrict__ Wq,
                         const float* __restrict__ Wv,
                         float* __restrict__ out) {
    extern __shared__ __align__(16) char smem[];
    _Float16* Xs  = (_Float16*)(smem + OFF_X);
    _Float16* Qs  = (_Float16*)(smem + OFF_Q);
    _Float16* Ks  = (_Float16*)(smem + OFF_K);
    _Float16* Vt  = (_Float16*)(smem + OFF_VT);
    float*    Sf  = (float*)   (smem + OFF_SF);
    _Float16* Sh  = (_Float16*)(smem + OFF_SH);

    const int tid  = threadIdx.x;
    const int lane = tid & 31;
    const int wave = tid >> 5;
    const int head = blockIdx.x;          // 256 heads
    const int b    = head >> 6;           // batch
    const int j    = head & 63;           // position within segment

    // ---- zero odd output tokens of this head's 128-row block (d_out is poisoned) ----
    {
        const size_t obase = ((size_t)b * TLEN + (size_t)128 * j) * DK;
        for (int i = tid; i < NSD * DK; i += 256) {
            int row = i >> 7, col = i & 127;
            out[obase + (size_t)(2 * row + 1) * DK + col] = 0.0f;
        }
    }

    // ---- phase 1: stage X (64 x 1024, rows are tokens t = 128*l + j), f32 -> f16 ----
    for (int i = tid; i < NSD * (EDIM / 4); i += 256) {
        int l = i >> 8;                 // 256 float4 per row
        int c = (i & 255) << 2;
        const size_t t = (size_t)b * TLEN + (size_t)128 * l + j;
        f4 v = *(const f4*)(x + t * EDIM + c);
        half4 h;
#pragma unroll
        for (int q = 0; q < 4; ++q) h[q] = (_Float16)v[q];
        *(half4*)(Xs + l * XP + c) = h;
    }
    __syncthreads();

    // ---- phase 2: Q/K/V projections. Wave w owns m-tile w of each matrix. ----
    // Q,K computed as D = W x X^T  (M=m, N=l): lane holds fixed l, contiguous m
    //   -> row-major Q/K writeback is one packed b128 store per tile.
    // V  computed as D = X x W^T  (M=l, N=m): lane holds fixed m, contiguous l
    //   -> transposed-V writeback is one packed b128 store per tile.
    {
        const int mb   = wave * 16;
        const int l_lo = lane & 15;
        const int sel8 = (lane >> 4) << 3;

        // ---- Q ----
        {
            v8f acc[4] = {};
            for (int kc = 0; kc < EDIM; kc += 32) {
                v16h aw = gblAW(Wq, mb, kc);           // converted once, used 4x
#pragma unroll
                for (int lt = 0; lt < 4; ++lt)
                    acc[lt] = WMMA_F16(aw, ldsB(Xs, XP, lt * 16, kc), acc[lt]);
            }
#pragma unroll
            for (int lt = 0; lt < 4; ++lt) {
                half8 pk;
#pragma unroll
                for (int g = 0; g < 8; ++g) pk[g] = (_Float16)acc[lt][g];
                *(half8*)(Qs + (lt * 16 + l_lo) * QP + mb + sel8) = pk;
            }
        }
        // ---- K ----
        {
            v8f acc[4] = {};
            for (int kc = 0; kc < EDIM; kc += 32) {
                v16h aw = gblAW(Wk, mb, kc);
#pragma unroll
                for (int lt = 0; lt < 4; ++lt)
                    acc[lt] = WMMA_F16(aw, ldsB(Xs, XP, lt * 16, kc), acc[lt]);
            }
#pragma unroll
            for (int lt = 0; lt < 4; ++lt) {
                half8 pk;
#pragma unroll
                for (int g = 0; g < 8; ++g) pk[g] = (_Float16)acc[lt][g];
                *(half8*)(Ks + (lt * 16 + l_lo) * QP + mb + sel8) = pk;
            }
        }
        // ---- V (transposed store) ----
        {
            v8f acc[4] = {};
            for (int kc = 0; kc < EDIM; kc += 32) {
                v16h bw = gblBW(Wv, mb, kc);
#pragma unroll
                for (int lt = 0; lt < 4; ++lt)
                    acc[lt] = WMMA_F16(ldsA(Xs, XP, lt * 16, kc), bw, acc[lt]);
            }
            const int m = mb + l_lo;
#pragma unroll
            for (int lt = 0; lt < 4; ++lt) {
                half8 pk;
#pragma unroll
                for (int g = 0; g < 8; ++g) pk[g] = (_Float16)acc[lt][g];
                *(half8*)(Vt + m * VP + lt * 16 + sel8) = pk;
            }
        }
    }
    __syncthreads();

    // ---- phase 3: S = K * Q^T (M=l, N=o), mask(o>l), +P[128l,128o], *norm; store S^T rows ----
    for (int t = wave; t < 16; t += 8) {
        const int mrow = t >> 2;            // l-tile
        const int ncol = t & 3;             // o-tile
        v8f acc = {};
        for (int kc = 0; kc < DK; kc += 32) {
            v16h a  = ldsA(Ks, QP, mrow * 16, kc);  // A = K rows l
            v16h bf = ldsB(Qs, QP, ncol * 16, kc);  // B[p][o] = Q[o][p]
            acc = WMMA_F16(a, bf, acc);
        }
        const int o  = ncol * 16 + (lane & 15);
        const int lb = mrow * 16 + ((lane >> 4) << 3);
        f4 r0, r1;
#pragma unroll
        for (int g = 0; g < 8; ++g) {
            const int l = lb + g;
            float v = acc[g];
            if (o > l) v = BIGNEG;                               // mask first
            v += P[(size_t)(128 * l) * TLEN + 128 * o];          // then bias
            v *= NORMF;                                          // then scale
            if (g < 4) r0[g] = v; else r1[g - 4] = v;
        }
        *(f4*)(Sf + o * SP + lb)     = r0;                       // store transposed: St[o][l]
        *(f4*)(Sf + o * SP + lb + 4) = r1;
    }
    __syncthreads();

    // ---- phase 4: softmax over l for each o (rows of St are contiguous) ----
    if (tid < NSD) {
        const float* r = Sf + tid * SP;
        float m = -3.4e38f;
        for (int i = 0; i < NSD; ++i) m = fmaxf(m, r[i]);
        float s = 0.0f;
        for (int i = 0; i < NSD; ++i) s += __expf(r[i] - m);
        const float inv = 1.0f / s;
        _Float16* d = Sh + tid * HP;
        for (int i = 0; i < NSD; ++i) d[i] = (_Float16)(__expf(r[i] - m) * inv);
    }
    __syncthreads();

    // ---- phase 5: att[d][m] = sum_l St[d][l] * Vt[m][l]; direct global store ----
    for (int t = wave; t < 32; t += 8) {
        const int drow = t >> 3;            // d-tile (0..3)
        const int mcol = t & 7;             // m-tile (0..7)
        v8f acc = {};
        for (int kc = 0; kc < NSD; kc += 32) {
            v16h a  = ldsA(Sh, HP, drow * 16, kc);  // A = St rows d (K = l)
            v16h bf = ldsB(Vt, VP, mcol * 16, kc);  // B[l][m] = Vt[m][l]
            acc = WMMA_F16(a, bf, acc);
        }
        const int m  = mcol * 16 + (lane & 15);
        const int db = drow * 16 + ((lane >> 4) << 3);
#pragma unroll
        for (int g = 0; g < 8; ++g) {
            const int d = db + g;           // output token = 128*j + 2*d
            out[((size_t)b * TLEN + (size_t)128 * j + 2 * d) * DK + m] = acc[g];
        }
    }
}

extern "C" void kernel_launch(void* const* d_in, const int* in_sizes, int n_in,
                              void* d_out, int out_size, void* d_ws, size_t ws_size,
                              hipStream_t stream) {
    (void)in_sizes; (void)n_in; (void)out_size; (void)d_ws; (void)ws_size;
    const float* x  = (const float*)d_in[0];
    const float* P  = (const float*)d_in[1];
    const float* Wk = (const float*)d_in[2];
    const float* Wq = (const float*)d_in[3];
    const float* Wv = (const float*)d_in[4];
    float* out = (float*)d_out;
    dim3 grid(BATCH * SEG);   // 256 heads
    dim3 block(256);          // 8 wave32s
    hipLaunchKernelGGL(dilated_attn_kernel, grid, block, LDS_BYTES, stream,
                       x, P, Wk, Wq, Wv, out);
}
